// GCN_Time_norm_34918084116704
// MI455X (gfx1250) — matrix operand order repaired and checked by reading, not compiled, not run on hardware
//
#include <hip/hip_runtime.h>
#include <hip/hip_bf16.h>

// ---------------- CDNA5 WMMA types ----------------
typedef __attribute__((ext_vector_type(16))) __bf16 v16bf;
typedef __attribute__((ext_vector_type(8)))  __bf16 v8bf;
typedef __attribute__((ext_vector_type(8)))  float  v8f;

static __device__ __forceinline__ unsigned short f2bf_bits(float f) {
  unsigned u = __builtin_bit_cast(unsigned, f);
  return (unsigned short)((u + 0x7FFFu + ((u >> 16) & 1u)) >> 16);  // RNE
}

namespace cfg {
constexpr int BG = 32;
constexpr int N0 = 1024;
constexpr int E  = 32 * 1024 * 8;             // 262144 edges, constant all layers
constexpr int K1 = 615, K2 = 369, K3 = 185;   // ceil(0.6*1024), ceil(0.6*615), ceil(0.5*369)
constexpr int n0 = BG * N0;                   // 32768
constexpr int n1 = BG * K1;                   // 19680
constexpr int n2 = BG * K2;                   // 11808
constexpr int n3 = BG * K3;                   // 5920
}

// ---------------- small utility kernels ----------------
__global__ void fillf_kernel(float* p, long n, float v) {
  long i = (long)blockIdx.x * blockDim.x + threadIdx.x;
  if (i < n) p[i] = v;
}

// flat f32 -> bf16 convert (layer-1 GEMM input)
__global__ void x2bf_kernel(const float* __restrict__ x, unsigned short* __restrict__ y, long n) {
  long i = (long)blockIdx.x * blockDim.x + threadIdx.x;
  if (i < n) y[i] = f2bf_bits(x[i]);
}

// Wt[n*K + k] = bf16(W[k*Nc + n])  (transpose + convert, cheap, once per launch)
__global__ void wt_kernel(const float* __restrict__ W, unsigned short* __restrict__ Wt,
                          int K, int Nc) {
  int i = blockIdx.x * blockDim.x + threadIdx.x;
  if (i >= K * Nc) return;
  int k = i / Nc, n = i - k * Nc;
  Wt[(size_t)n * K + k] = f2bf_bits(W[i]);
}

// ---------------- WMMA GEMM: C[M,Nc] = A[M,K] @ W[K,Nc] ----------------
// A: bf16 row-major [M,K]. Wt: bf16 pre-transposed [Nc,K]. C: f32 row-major.
// block = 256 threads = 8 waves arranged 2(row) x 4(colgroup); each wave does a
// 16x64 tile via 4 accumulators (A fragment reused across 4 WMMAs per K-step).
// Block tile = 32 x 256. M % 32 == 0 and Nc % 256 == 0 for every layer here.
__global__ void __launch_bounds__(256)
gemm_bf16_wmma(const unsigned short* __restrict__ A, const unsigned short* __restrict__ Wt,
               float* __restrict__ C, int M, int K, int Nc) {
  const int lane = threadIdx.x & 31;
  const int wave = threadIdx.x >> 5;
  const int m0 = blockIdx.x * 32 + (wave >> 2) * 16;
  const int nb = blockIdx.y * 256 + (wave & 3) * 64;
  const int r  = lane & 15;
  const int hi = lane >> 4;

  v8f acc0 = {0.f,0.f,0.f,0.f,0.f,0.f,0.f,0.f};
  v8f acc1 = acc0, acc2 = acc0, acc3 = acc0;

  const unsigned short* arow = A + (size_t)(m0 + r) * K;
  const unsigned short* b0p = Wt + (size_t)(nb +  0 + r) * K + hi * 16;
  const unsigned short* b1p = Wt + (size_t)(nb + 16 + r) * K + hi * 16;
  const unsigned short* b2p = Wt + (size_t)(nb + 32 + r) * K + hi * 16;
  const unsigned short* b3p = Wt + (size_t)(nb + 48 + r) * K + hi * 16;

  for (int k0 = 0; k0 < K; k0 += 32) {
    // A fragment (16x32 bf16): lane(hi,r) holds row m0+r, K = k0+hi*16+{0..15} contiguous
    v8bf alo = *reinterpret_cast<const v8bf*>(arow + k0 + hi * 16);
    v8bf ahi = *reinterpret_cast<const v8bf*>(arow + k0 + hi * 16 + 8);
    v16bf a = __builtin_shufflevector(alo, ahi, 0,1,2,3,4,5,6,7,8,9,10,11,12,13,14,15);
    // B fragments (32x16 bf16): lane(hi,r) holds col, K = k0+hi*16+{0..15} contiguous
    v16bf b0 = *reinterpret_cast<const v16bf*>(b0p + k0);
    v16bf b1 = *reinterpret_cast<const v16bf*>(b1p + k0);
    v16bf b2 = *reinterpret_cast<const v16bf*>(b2p + k0);
    v16bf b3 = *reinterpret_cast<const v16bf*>(b3p + k0);
    acc0 = __builtin_amdgcn_wmma_f32_16x16x32_bf16(false, a, false, b0, (short)0, acc0, false, false);
    acc1 = __builtin_amdgcn_wmma_f32_16x16x32_bf16(false, a, false, b1, (short)0, acc1, false, false);
    acc2 = __builtin_amdgcn_wmma_f32_16x16x32_bf16(false, a, false, b2, (short)0, acc2, false, false);
    acc3 = __builtin_amdgcn_wmma_f32_16x16x32_bf16(false, a, false, b3, (short)0, acc3, false, false);
  }
  // C/D layout: VGPR v -> row m0 + hi*8 + v, col = tile_base + r
  float* crow = C + (size_t)(m0 + hi * 8) * Nc + nb + r;
#pragma unroll
  for (int v = 0; v < 8; ++v) {
    float* cr = crow + (size_t)v * Nc;
    cr[0]  = acc0[v];
    cr[16] = acc1[v];
    cr[32] = acc2[v];
    cr[48] = acc3[v];
  }
}

// ---------------- graph kernels ----------------
__global__ void edge_deg_kernel(const int* __restrict__ dst, const float* __restrict__ ew,
                                float* __restrict__ deg, int E_) {
  int e = blockIdx.x * blockDim.x + threadIdx.x;
  if (e >= E_) return;
  float w = ew[e];
  if (w != 0.f) atomicAdd(deg + dst[e], w);
}

__global__ void rsqrt_deg_kernel(const float* __restrict__ deg, float* __restrict__ dis, int n) {
  int i = blockIdx.x * blockDim.x + threadIdx.x;
  if (i >= n) return;
  float d = deg[i];
  dis[i] = (d > 0.f) ? rsqrtf(fmaxf(d, 1e-12f)) : 0.f;
}

__global__ void fill_bias_kernel(float* __restrict__ out, const float* __restrict__ bias,
                                 long total, int C) {
  long i = (long)blockIdx.x * blockDim.x + threadIdx.x;
  if (i < total) out[i] = bias[i % C];
}

// out[dst] += h[src] * dis[src]*dis[dst]*ew, float4-wide f32 atomics (L2-resident)
__global__ void aggregate_kernel(const float* __restrict__ h, const int* __restrict__ src,
                                 const int* __restrict__ dst, const float* __restrict__ ew,
                                 const float* __restrict__ dis, float* __restrict__ out,
                                 int E_, int C) {
  long t = (long)blockIdx.x * blockDim.x + threadIdx.x;
  int cq = C >> 2;
  long e = t / cq;
  if (e >= E_) return;
  float w = ew[e];
  if (w == 0.f) return;                 // masked edge -> no work
  int s = src[e], d = dst[e];
  w *= dis[s] * dis[d];
  if (w == 0.f) return;
  int c4 = (int)(t - e * cq) << 2;
  float4 hv = *reinterpret_cast<const float4*>(h + (size_t)s * C + c4);
  float* o = out + (size_t)d * C + c4;
  atomicAdd(o + 0, hv.x * w);
  atomicAdd(o + 1, hv.y * w);
  atomicAdd(o + 2, hv.z * w);
  atomicAdd(o + 3, hv.w * w);
}

// one block per channel: mean + biased var over n rows
__global__ void bn_stats_kernel(const float* __restrict__ x, int n, int C,
                                float* __restrict__ mean, float* __restrict__ var) {
  __shared__ float s1[256], s2[256];
  int c = blockIdx.x, t = threadIdx.x;
  float a = 0.f, b = 0.f;
  for (int r = t; r < n; r += 256) {
    float v = x[(size_t)r * C + c];
    a += v; b += v * v;
  }
  s1[t] = a; s2[t] = b; __syncthreads();
  for (int off = 128; off; off >>= 1) {
    if (t < off) { s1[t] += s1[t + off]; s2[t] += s2[t + off]; }
    __syncthreads();
  }
  if (t == 0) {
    float m = s1[0] / (float)n;
    mean[c] = m;
    var[c]  = s2[0] / (float)n - m * m;
  }
}

__global__ void bn_norm_tanh_kernel(float* __restrict__ x, long total, int C,
                                    const float* __restrict__ mean, const float* __restrict__ var,
                                    const float* __restrict__ g, const float* __restrict__ be) {
  long i = (long)blockIdx.x * blockDim.x + threadIdx.x;
  if (i >= total) return;
  int c = (int)(i % C);
  float v = (x[i] - mean[c]) * rsqrtf(var[c] + 1e-5f) * g[c] + be[c];
  x[i] = tanhf(v);
}

// per node: srel = x . rel_w ; score = rel_b + x . root_w   (one wave32 per node)
__global__ void score_dots_kernel(const float* __restrict__ x, int n, int C,
                                  const float* __restrict__ relw, const float* __restrict__ rootw,
                                  const float* __restrict__ relb,
                                  float* __restrict__ srel, float* __restrict__ score) {
  int lane = threadIdx.x & 31;
  int node = blockIdx.x * 8 + (threadIdx.x >> 5);
  if (node >= n) return;
  const float* xr = x + (size_t)node * C;
  float a = 0.f, b = 0.f;
  for (int c = lane; c < C; c += 32) {
    float v = xr[c];
    a += v * relw[c];
    b += v * rootw[c];
  }
  for (int off = 16; off; off >>= 1) {
    a += __shfl_xor(a, off);
    b += __shfl_xor(b, off);
  }
  if (lane == 0) { srel[node] = a; score[node] = relb[0] + b; }
}

__global__ void edge_score_kernel(const int* __restrict__ src, const int* __restrict__ dst,
                                  const float* __restrict__ ew, const float* __restrict__ srel,
                                  float* __restrict__ score, int E_) {
  int e = blockIdx.x * blockDim.x + threadIdx.x;
  if (e >= E_) return;
  float w = ew[e];
  if (w != 0.f) atomicAdd(score + dst[e], srel[src[e]] * w);
}

// per-graph descending bitonic top-k (tie-break: lower index first, matches jax top_k)
__global__ void __launch_bounds__(1024)
topk_sort_kernel(const float* __restrict__ score, int nper, int k,
                 int* __restrict__ perm, int* __restrict__ newid) {
  __shared__ float ssc[1024];
  __shared__ int   sidx[1024];
  int b = blockIdx.x, tid = threadIdx.x;
  ssc[tid]  = (tid < nper) ? score[b * nper + tid] : -3.4e38f;
  sidx[tid] = tid;
  __syncthreads();
  for (int ks = 2; ks <= 1024; ks <<= 1) {
    for (int j = ks >> 1; j > 0; j >>= 1) {
      int ixj = tid ^ j;
      if (ixj > tid) {
        bool dir = ((tid & ks) == 0);  // descending in "dir" subsequences
        float si = ssc[tid], sj = ssc[ixj];
        int   ii = sidx[tid], ij = sidx[ixj];
        bool before = (si > sj) || (si == sj && ii < ij);
        if (dir ? !before : before) {
          ssc[tid] = sj; ssc[ixj] = si;
          sidx[tid] = ij; sidx[ixj] = ii;
        }
      }
      __syncthreads();
    }
  }
  int li = sidx[tid];
  if (li < nper) {
    int g = b * nper + li;
    if (tid < k) {
      newid[g] = b * k + tid;
      perm[b * k + tid] = g;
    } else {
      newid[g] = -1;
    }
  }
}

// gather + gate, writing bf16 directly (consumed only by next layer's WMMA GEMM)
__global__ void xnew_bf_kernel(const float* __restrict__ x, const int* __restrict__ perm,
                               const float* __restrict__ score, unsigned short* __restrict__ y,
                               int C) {
  int j = blockIdx.x;
  int p = perm[j];
  float t = tanhf(score[p]);
  const float* xr = x + (size_t)p * C;
  unsigned short* yr = y + (size_t)j * C;
  for (int c = threadIdx.x; c < C; c += blockDim.x) yr[c] = f2bf_bits(xr[c] * t);
}

__global__ void relabel_kernel(const int* __restrict__ src, const int* __restrict__ dst,
                               const float* __restrict__ ew, const int* __restrict__ newid,
                               int* __restrict__ nsrc, int* __restrict__ ndst,
                               float* __restrict__ new_ew, int E_) {
  int e = blockIdx.x * blockDim.x + threadIdx.x;
  if (e >= E_) return;
  int s = newid[src[e]], d = newid[dst[e]];
  bool keep = (s >= 0) && (d >= 0);
  nsrc[e]  = keep ? s : 0;
  ndst[e]  = keep ? d : 0;
  new_ew[e] = keep ? ew[e] : 0.f;
}

__global__ void readout_kernel(const float* __restrict__ h, float* __restrict__ out, int k3) {
  int b = blockIdx.x, c = threadIdx.x;  // 256 threads, C = 256
  const float* p = h + (size_t)b * k3 * 256 + c;
  float mx = -3.4e38f, sm = 0.f;
  for (int r = 0; r < k3; ++r) {
    float v = p[(size_t)r * 256];
    mx = fmaxf(mx, v);
    sm += v;
  }
  out[b * 512 + c]       = mx;
  out[b * 512 + 256 + c] = sm / (float)k3;
}

// ---------------- host orchestration ----------------
extern "C" void kernel_launch(void* const* d_in, const int* in_sizes, int n_in,
                              void* d_out, int out_size, void* d_ws, size_t ws_size,
                              hipStream_t stream) {
  using namespace cfg;
  const float* X0  = (const float*)d_in[0];
  const int*   EI  = (const int*)d_in[1];
  const float* W1  = (const float*)d_in[2];  const float* b1  = (const float*)d_in[3];
  const float* W2  = (const float*)d_in[4];  const float* b2  = (const float*)d_in[5];
  const float* W3  = (const float*)d_in[6];  const float* b3  = (const float*)d_in[7];
  const float* W4  = (const float*)d_in[8];  const float* b4  = (const float*)d_in[9];
  const float* g1  = (const float*)d_in[10]; const float* be1 = (const float*)d_in[11];
  const float* g2  = (const float*)d_in[12]; const float* be2 = (const float*)d_in[13];
  const float* g3  = (const float*)d_in[14]; const float* be3 = (const float*)d_in[15];
  const float* p1rw = (const float*)d_in[16]; const float* p1rb = (const float*)d_in[17];
  const float* p1ow = (const float*)d_in[18];
  const float* p2rw = (const float*)d_in[19]; const float* p2rb = (const float*)d_in[20];
  const float* p2ow = (const float*)d_in[21];
  const float* p3rw = (const float*)d_in[22]; const float* p3rb = (const float*)d_in[23];
  const float* p3ow = (const float*)d_in[24];
  float* out = (float*)d_out;

  const int* src0 = EI;
  const int* dst0 = EI + E;

  // workspace carve-out (~168 MB)
  char* wp = (char*)d_ws;
  auto take = [&](size_t bytes) -> void* {
    void* p = (void*)wp;
    wp += (bytes + 255) & ~(size_t)255;
    return p;
  };
  float* H   = (float*)take((size_t)n0 * 512 * 4);            // GEMM output
  float* G   = (float*)take((size_t)n0 * 512 * 4);            // aggregate / BN / pool buffer
  unsigned short* Xbf = (unsigned short*)take((size_t)n0 * 512 * 2);  // bf16 GEMM input
  unsigned short* Wt1 = (unsigned short*)take((size_t)512 * 512 * 2);
  unsigned short* Wt2 = (unsigned short*)take((size_t)512 * 512 * 2);
  unsigned short* Wt3 = (unsigned short*)take((size_t)256 * 512 * 2);
  unsigned short* Wt4 = (unsigned short*)take((size_t)256 * 256 * 2);
  int*   esA = (int*)take((size_t)E * 4);
  int*   edA = (int*)take((size_t)E * 4);
  float* ewA = (float*)take((size_t)E * 4);
  int*   esB = (int*)take((size_t)E * 4);
  int*   edB = (int*)take((size_t)E * 4);
  float* ewB = (float*)take((size_t)E * 4);
  float* deg   = (float*)take((size_t)n0 * 4);
  float* dis   = (float*)take((size_t)n0 * 4);
  float* score = (float*)take((size_t)n0 * 4);
  float* srel  = (float*)take((size_t)n0 * 4);
  int*   newid = (int*)take((size_t)n0 * 4);
  int*   perm  = (int*)take((size_t)n1 * 4);
  float* mean  = (float*)take(512 * 4);
  float* var   = (float*)take(512 * 4);
  (void)ws_size; (void)in_sizes; (void)n_in; (void)out_size;

  // weights -> transposed bf16; layer-1 features -> bf16; layer-1 edge weights = 1
  wt_kernel<<<(512 * 512 + 255) / 256, 256, 0, stream>>>(W1, Wt1, 512, 512);
  wt_kernel<<<(512 * 512 + 255) / 256, 256, 0, stream>>>(W2, Wt2, 512, 512);
  wt_kernel<<<(512 * 256 + 255) / 256, 256, 0, stream>>>(W3, Wt3, 512, 256);
  wt_kernel<<<(256 * 256 + 255) / 256, 256, 0, stream>>>(W4, Wt4, 256, 256);
  {
    long tot = (long)n0 * 512;
    x2bf_kernel<<<(unsigned)((tot + 255) / 256), 256, 0, stream>>>(X0, Xbf, tot);
  }
  fillf_kernel<<<(E + 255) / 256, 256, 0, stream>>>(ewA, E, 1.0f);

  auto conv = [&](const unsigned short* Xb, const unsigned short* Wt, const float* bias,
                  int n, int K, int C,
                  const int* src, const int* dst, const float* ew) {
    dim3 grid(n / 32, C / 256);
    gemm_bf16_wmma<<<grid, 256, 0, stream>>>(Xb, Wt, H, n, K, C);
    fillf_kernel<<<(n + 255) / 256, 256, 0, stream>>>(deg, n, 0.f);
    edge_deg_kernel<<<(E + 255) / 256, 256, 0, stream>>>(dst, ew, deg, E);
    rsqrt_deg_kernel<<<(n + 255) / 256, 256, 0, stream>>>(deg, dis, n);
    long total = (long)n * C;
    fill_bias_kernel<<<(unsigned)((total + 255) / 256), 256, 0, stream>>>(G, bias, total, C);
    long work = (long)E * (C >> 2);
    aggregate_kernel<<<(unsigned)((work + 255) / 256), 256, 0, stream>>>(H, src, dst, ew, dis, G, E, C);
  };
  auto bntanh = [&](int n, int C, const float* g, const float* be) {
    bn_stats_kernel<<<C, 256, 0, stream>>>(G, n, C, mean, var);
    long total = (long)n * C;
    bn_norm_tanh_kernel<<<(unsigned)((total + 255) / 256), 256, 0, stream>>>(G, total, C, mean, var, g, be);
  };
  auto pool = [&](int n, int C, int nper, int k,
                  const float* rw, const float* rb, const float* ow,
                  const int* src, const int* dst, const float* ew,
                  int* nsrc, int* ndst, float* new_ew) {
    score_dots_kernel<<<(n + 7) / 8, 256, 0, stream>>>(G, n, C, rw, ow, rb, srel, score);
    edge_score_kernel<<<(E + 255) / 256, 256, 0, stream>>>(src, dst, ew, srel, score, E);
    topk_sort_kernel<<<BG, 1024, 0, stream>>>(score, nper, k, perm, newid);
    xnew_bf_kernel<<<BG * k, 256, 0, stream>>>(G, perm, score, Xbf, C);  // bf16 out
    relabel_kernel<<<(E + 255) / 256, 256, 0, stream>>>(src, dst, ew, newid, nsrc, ndst, new_ew, E);
  };

  // ---- Layer 1
  conv(Xbf, Wt1, b1, n0, 512, 512, src0, dst0, ewA);
  bntanh(n0, 512, g1, be1);
  pool(n0, 512, N0, K1, p1rw, p1rb, p1ow, src0, dst0, ewA, esB, edB, ewB);

  // ---- Layer 2
  conv(Xbf, Wt2, b2, n1, 512, 512, esB, edB, ewB);
  bntanh(n1, 512, g2, be2);
  pool(n1, 512, K1, K2, p2rw, p2rb, p2ow, esB, edB, ewB, esA, edA, ewA);

  // ---- Layer 3
  conv(Xbf, Wt3, b3, n2, 512, 256, esA, edA, ewA);
  bntanh(n2, 256, g3, be3);
  pool(n2, 256, K2, K3, p3rw, p3rb, p3ow, esA, edA, ewA, esB, edB, ewB);

  // ---- Layer 4 + readout
  conv(Xbf, Wt4, b4, n3, 256, 256, esB, edB, ewB);
  readout_kernel<<<BG, 256, 0, stream>>>(G, out, K3);
}
